// MLPGate3_16149077033389
// MI455X (gfx1250) — matrix operand order repaired and checked by reading, not compiled, not run on hardware
//
#include <hip/hip_runtime.h>
#include <math.h>

#define H2 128
#define SE 256   // enumerated states

typedef _Float16 f16;
typedef __attribute__((ext_vector_type(16))) _Float16 v16h;
typedef __attribute__((ext_vector_type(8)))  _Float16 v8h;
typedef __attribute__((ext_vector_type(8)))  float    v8f;

// ---------------------------------------------------------------------------
// WMMA helpers (wave32, V_WMMA_F32_16X16X32_F16)
// ---------------------------------------------------------------------------

// Wave-local LDS ordering: LDS ops are in-order per wave; this forces the
// compiler not to reorder and waits for outstanding DS ops.
__device__ __forceinline__ void lds_fence_wave() {
  asm volatile("s_wait_dscnt 0" ::: "memory");
}

// A-fragment (16x32 f16) from a row-major LDS tile [16][ld].
// ISA layout: lanes 0-15 hold M=lane, K={0..7,16..23}; lanes 16-31 K={8..15,24..31}.
__device__ __forceinline__ v16h frag_a_lds(const f16* tile, int ld, int k0, int lane) {
  int m  = lane & 15;
  int kb = k0 + ((lane >> 4) << 3);
  const f16* p = tile + m * ld + kb;
  v8h lo = *(const v8h*)(p);
  v8h hi = *(const v8h*)(p + 16);
  v16h a;
#pragma unroll
  for (int i = 0; i < 8; ++i) { a[i] = lo[i]; a[8 + i] = hi[i]; }
  return a;
}

// B-fragment (32x16 f16) from global weights stored transposed [out][in] (ld=in).
// ISA layout: lanes 0-15 hold N=lane, K={0..7,16..23}; lanes 16-31 K={8..15,24..31}.
__device__ __forceinline__ v16h frag_b_glb(const f16* wt, int ld, int n0, int k0, int lane) {
  int n  = lane & 15;
  int kb = k0 + ((lane >> 4) << 3);
  const f16* p = wt + (size_t)(n0 + n) * ld + kb;
  v8h lo = *(const v8h*)(p);
  v8h hi = *(const v8h*)(p + 16);
  v16h b;
#pragma unroll
  for (int i = 0; i < 8; ++i) { b[i] = lo[i]; b[8 + i] = hi[i]; }
  return b;
}

#define WMMA_F16(a, b, c) \
  __builtin_amdgcn_wmma_f32_16x16x32_f16(false, (a), false, (b), (short)0, (c), false, false)

// ---------------------------------------------------------------------------
// Bucketing (sort edges/nodes by gate type) + weight conversion + utilities
// ---------------------------------------------------------------------------

__global__ void zero_ints_kernel(int* p, int n) {
  int i = blockIdx.x * blockDim.x + threadIdx.x;
  if (i < n) p[i] = 0;
}

__global__ void zero_floats_kernel(float* p, long long n) {
  long long i = (long long)blockIdx.x * blockDim.x + threadIdx.x;
  if (i < n) p[i] = 0.0f;
}

__global__ void count_types_kernel(const int* keyidx, const int* gtype, int n, int* cnt) {
  int i = blockIdx.x * blockDim.x + threadIdx.x;
  if (i < n) {
    int node = keyidx ? keyidx[i] : i;
    atomicAdd(&cnt[gtype[node]], 1);
  }
}

__global__ void make_offsets_kernel(const int* cnt, int* cur) {
  cur[0] = 0;
  cur[1] = cnt[0];
  cur[2] = cnt[0] + cnt[1];
}

__global__ void scatter_sorted_kernel(const int* keyidx, const int* gtype, int n,
                                      int* cur, int* out) {
  int i = blockIdx.x * blockDim.x + threadIdx.x;
  if (i < n) {
    int node = keyidx ? keyidx[i] : i;
    int t = gtype[node];
    int pos = atomicAdd(&cur[t], 1);
    out[pos] = i;
  }
}

// src (R,C) row-major -> dst (C,R) row-major f16  (i.e. [out][in])
__global__ void convert_transpose_kernel(const float* src, f16* dst, int R, int C) {
  int i = blockIdx.x * blockDim.x + threadIdx.x;
  if (i < R * C) {
    int r = i / C, c = i % C;
    dst[(size_t)c * R + r] = (f16)src[i];
  }
}

__global__ void convert_plain_kernel(const float* src, f16* dst, int n) {
  int i = blockIdx.x * blockDim.x + threadIdx.x;
  if (i < n) dst[i] = (f16)src[i];
}

// ---------------------------------------------------------------------------
// Edge message MLP + atomic segment-sum (one branch; per-dst-type weights)
// ---------------------------------------------------------------------------

struct BranchW {
  const f16*   w1[3]; const f16*   w2[3]; const f16*   w3[3];   // [out][in] f16
  const float* b1[3]; const float* b2[3]; const float* b3[3];
};

#define EW_WAVES 2

__global__ __launch_bounds__(64) void edge_mlp_kernel(
    const int* __restrict__ sorted_e, int E,
    const int* __restrict__ esrc, const int* __restrict__ edst,
    const int* __restrict__ gtype,
    const float* __restrict__ hs, const float* __restrict__ hf,
    const float* __restrict__ hq,
    int Kin, BranchW W, float* __restrict__ agg)
{
  __shared__ __align__(16) f16 smsg[EW_WAVES][16 * 384];
  __shared__ __align__(16) f16 sact1[EW_WAVES][16 * 128];
  __shared__ __align__(16) f16 sact2[EW_WAVES][16 * 128];
  __shared__ int rdst[EW_WAVES][16];
  __shared__ int rtyp[EW_WAVES][16];

  const int wave = threadIdx.x >> 5;
  const int lane = threadIdx.x & 31;
  const int tile = blockIdx.x * EW_WAVES + wave;
  const int e0 = tile * 16;
  if (e0 >= E) return;

  f16* msg  = smsg[wave];
  f16* act1 = sact1[wave];
  f16* act2 = sact2[wave];

  if (lane < 16) {
    int ei = e0 + lane;
    int t = -1, d = 0;
    if (ei < E) { int e = sorted_e[ei]; d = edst[e]; t = gtype[d]; }
    rdst[wave][lane] = d;
    rtyp[wave][lane] = t;
  }
  for (int r = 0; r < 16; ++r) {
    int ei = e0 + r;
    int e  = sorted_e[ei < E ? ei : (E - 1)];
    int s  = esrc[e];
    const float* ps = hs + (size_t)s * H2;
    const float* pf = hf + (size_t)s * H2;
    const float* pq = hq + (size_t)s * H2;
    for (int k = lane; k < Kin; k += 32) {
      float v = (k < 128) ? ps[k] : (k < 256) ? pf[k - 128] : pq[k - 256];
      msg[r * Kin + k] = (f16)v;
    }
  }
  lds_fence_wave();

  int tmin = 3, tmax = -1;
  for (int r = 0; r < 16; ++r) {
    int t = rtyp[wave][r];
    if (t >= 0) { tmin = t < tmin ? t : tmin; tmax = t > tmax ? t : tmax; }
  }

  const int nlo = lane & 15;
  const int mh  = (lane >> 4) << 3;

  for (int t = tmin; t <= tmax; ++t) {   // wave-uniform loop (<=2 types per tile)
    // layer 1: msg(16xKin) @ W1 -> act1(16x128), +bias, relu
    for (int nt = 0; nt < 8; ++nt) {
      v8f acc = {};
      for (int kc = 0; kc < Kin; kc += 32)
        acc = WMMA_F16(frag_a_lds(msg, Kin, kc, lane),
                       frag_b_glb(W.w1[t], Kin, nt * 16, kc, lane), acc);
      int n = nt * 16 + nlo;
      float bias = W.b1[t][n];
#pragma unroll
      for (int r = 0; r < 8; ++r)
        act1[(mh + r) * 128 + n] = (f16)fmaxf(acc[r] + bias, 0.0f);
    }
    lds_fence_wave();
    // layer 2
    for (int nt = 0; nt < 8; ++nt) {
      v8f acc = {};
      for (int kc = 0; kc < 128; kc += 32)
        acc = WMMA_F16(frag_a_lds(act1, 128, kc, lane),
                       frag_b_glb(W.w2[t], 128, nt * 16, kc, lane), acc);
      int n = nt * 16 + nlo;
      float bias = W.b2[t][n];
#pragma unroll
      for (int r = 0; r < 8; ++r)
        act2[(mh + r) * 128 + n] = (f16)fmaxf(acc[r] + bias, 0.0f);
    }
    lds_fence_wave();
    // layer 3 + masked atomic scatter-add (mask applied to OUTPUT, like em)
    for (int nt = 0; nt < 8; ++nt) {
      v8f acc = {};
      for (int kc = 0; kc < 128; kc += 32)
        acc = WMMA_F16(frag_a_lds(act2, 128, kc, lane),
                       frag_b_glb(W.w3[t], 128, nt * 16, kc, lane), acc);
      int n = nt * 16 + nlo;
      float bias = W.b3[t][n];
#pragma unroll
      for (int r = 0; r < 8; ++r) {
        int m = mh + r;
        if (rtyp[wave][m] == t)
          atomicAdd(agg + (size_t)rdst[wave][m] * H2 + n, acc[r] + bias);
      }
    }
    lds_fence_wave();   // before next type overwrites act1/act2
  }
}

// ---------------------------------------------------------------------------
// GRU update (one branch; per-node-type weights), in-place on working h
// ---------------------------------------------------------------------------

struct GruW {
  const f16*   wih[3]; const f16*   whh[3];   // [384][128] f16 (already [out][in])
  const float* bih[3]; const float* bhh[3];
};

#define GR_WAVES 4

__global__ __launch_bounds__(128) void gru_kernel(
    const int* __restrict__ sorted_n, int N,
    const int* __restrict__ gtype,
    const float* __restrict__ agg, float* __restrict__ h, GruW W)
{
  __shared__ __align__(16) f16 sx[GR_WAVES][16 * 128];
  __shared__ __align__(16) f16 sh[GR_WAVES][16 * 128];
  __shared__ int rnode[GR_WAVES][16];
  __shared__ int rtyp[GR_WAVES][16];

  const int wave = threadIdx.x >> 5;
  const int lane = threadIdx.x & 31;
  const int tile = blockIdx.x * GR_WAVES + wave;
  const int n0 = tile * 16;
  if (n0 >= N) return;

  f16* xs = sx[wave];
  f16* hv = sh[wave];

  if (lane < 16) {
    int ni = n0 + lane;
    int node = 0, t = -1;
    if (ni < N) { node = sorted_n[ni]; t = gtype[node]; }
    rnode[wave][lane] = node;
    rtyp[wave][lane] = t;
  }
  for (int r = 0; r < 16; ++r) {
    int ni = n0 + r;
    int node = sorted_n[ni < N ? ni : (N - 1)];
    const float* px = agg + (size_t)node * H2;
    const float* ph = h   + (size_t)node * H2;
    for (int k = lane; k < 128; k += 32) {
      xs[r * 128 + k] = (f16)px[k];
      hv[r * 128 + k] = (f16)ph[k];
    }
  }
  lds_fence_wave();

  int tmin = 3, tmax = -1;
  for (int r = 0; r < 16; ++r) {
    int t = rtyp[wave][r];
    if (t >= 0) { tmin = t < tmin ? t : tmin; tmax = t > tmax ? t : tmax; }
  }

  const int nlo = lane & 15;
  const int mh  = (lane >> 4) << 3;

  for (int t = tmin; t <= tmax; ++t) {
    for (int nb = 0; nb < 8; ++nb) {
      int j0 = nb * 16;
      v8f a_ir = {}, a_iz = {}, a_in = {}, a_hr = {}, a_hz = {}, a_hn = {};
      for (int kc = 0; kc < 128; kc += 32) {
        v16h ax = frag_a_lds(xs, 128, kc, lane);
        v16h ah = frag_a_lds(hv, 128, kc, lane);
        a_ir = WMMA_F16(ax, frag_b_glb(W.wih[t], 128,       j0, kc, lane), a_ir);
        a_iz = WMMA_F16(ax, frag_b_glb(W.wih[t], 128, 128 + j0, kc, lane), a_iz);
        a_in = WMMA_F16(ax, frag_b_glb(W.wih[t], 128, 256 + j0, kc, lane), a_in);
        a_hr = WMMA_F16(ah, frag_b_glb(W.whh[t], 128,       j0, kc, lane), a_hr);
        a_hz = WMMA_F16(ah, frag_b_glb(W.whh[t], 128, 128 + j0, kc, lane), a_hz);
        a_hn = WMMA_F16(ah, frag_b_glb(W.whh[t], 128, 256 + j0, kc, lane), a_hn);
      }
      int j = j0 + nlo;
      float bir = W.bih[t][j], biz = W.bih[t][128 + j], bin = W.bih[t][256 + j];
      float bhr = W.bhh[t][j], bhz = W.bhh[t][128 + j], bhn = W.bhh[t][256 + j];
#pragma unroll
      for (int r = 0; r < 8; ++r) {
        int m = mh + r;
        if (rtyp[wave][m] == t) {
          int node = rnode[wave][m];
          float hold = h[(size_t)node * H2 + j];
          float rr = 1.0f / (1.0f + expf(-(a_ir[r] + bir + a_hr[r] + bhr)));
          float zz = 1.0f / (1.0f + expf(-(a_iz[r] + biz + a_hz[r] + bhz)));
          float nn = tanhf(a_in[r] + bin + rr * (a_hn[r] + bhn));
          h[(size_t)node * H2 + j] = (1.0f - zz) * nn + zz * hold;
        }
      }
    }
  }
}

// ---------------------------------------------------------------------------
// Readout: gather flip-flop states
// ---------------------------------------------------------------------------

__global__ void gather_ff_kernel(const float* hs, const float* hf, const float* hq,
                                 const int* ff, int Mff,
                                 float* h0, float* h1, float* h0t, float* h1t) {
  int idx = blockIdx.x * blockDim.x + threadIdx.x;
  if (idx >= Mff * H2) return;
  int m = idx >> 7, c = idx & 127;
  int node = ff[m];
  float vf = hf[(size_t)node * H2 + c];
  float vq = hq[(size_t)node * H2 + c];
  float vs = hs[(size_t)node * H2 + c];
  h0[idx] = vf; h1[idx] = vq; h0t[idx] = vs; h1t[idx] = 0.5f * (vf + vq);
}

// ---------------------------------------------------------------------------
// Single-head attention pool + mean over Mff rows (small; VALU f32).
// One block per state s.
// ---------------------------------------------------------------------------

__global__ __launch_bounds__(128) void pool_kernel(
    const float* __restrict__ a0, const float* __restrict__ a1, int Mff,
    const float* __restrict__ Wq, const float* __restrict__ Wk,
    const float* __restrict__ Wv, const float* __restrict__ Wo,
    float* __restrict__ g)
{
  __shared__ float sel[16 * 128];
  __shared__ float qb[16 * 128];
  __shared__ float kb[16 * 128];
  __shared__ float vb[16 * 128];
  __shared__ float att[16 * 16];
  const int s = blockIdx.x, tid = threadIdx.x;

  for (int idx = tid; idx < Mff * 128; idx += 128) {
    int m = idx >> 7;
    int bit = (s >> (Mff - 1 - m)) & 1;
    sel[idx] = bit ? a1[idx] : a0[idx];
  }
  __syncthreads();
  {
    int j = tid;  // 128 threads == 128 columns
    for (int m = 0; m < Mff; ++m) {
      float aq = 0.f, ak = 0.f, av = 0.f;
      for (int c = 0; c < 128; ++c) {
        float x = sel[m * 128 + c];
        aq += x * Wq[c * 128 + j];
        ak += x * Wk[c * 128 + j];
        av += x * Wv[c * 128 + j];
      }
      qb[m * 128 + j] = aq; kb[m * 128 + j] = ak; vb[m * 128 + j] = av;
    }
  }
  __syncthreads();
  if (tid < Mff * Mff) {
    int m = tid / Mff, n2 = tid % Mff;
    float d = 0.f;
    for (int c = 0; c < 128; ++c) d += qb[m * 128 + c] * kb[n2 * 128 + c];
    att[m * 16 + n2] = d * 0.0883883476483184f;   // 1/sqrt(128)
  }
  __syncthreads();
  if (tid < Mff) {
    float mx = -1e30f;
    for (int n2 = 0; n2 < Mff; ++n2) mx = fmaxf(mx, att[tid * 16 + n2]);
    float sum = 0.f;
    for (int n2 = 0; n2 < Mff; ++n2) {
      float e = expf(att[tid * 16 + n2] - mx);
      att[tid * 16 + n2] = e; sum += e;
    }
    float inv = 1.0f / sum;
    for (int n2 = 0; n2 < Mff; ++n2) att[tid * 16 + n2] *= inv;
  }
  __syncthreads();
  {
    int j = tid;
    float o[16];
    for (int m = 0; m < Mff; ++m) {
      float a = 0.f;
      for (int n2 = 0; n2 < Mff; ++n2) a += att[m * 16 + n2] * vb[n2 * 128 + j];
      o[m] = a;
    }
    __syncthreads();                       // sel reads all done; reuse as o
    for (int m = 0; m < Mff; ++m) sel[m * 128 + j] = o[m];
  }
  __syncthreads();
  {
    int j = tid;
    float acc = 0.f;
    for (int m = 0; m < Mff; ++m) {
      float a = 0.f;
      for (int c = 0; c < 128; ++c) a += sel[m * 128 + c] * Wo[c * 128 + j];
      acc += a;
    }
    g[(size_t)s * 128 + j] = acc / (float)Mff;
  }
}

// ---------------------------------------------------------------------------
// Readout MLP over S*S rows of [gq[j] | gi[i]]: 256 -> 128 -> 128 -> 1 (WMMA)
// ---------------------------------------------------------------------------

#define RO_WAVES 2

__global__ __launch_bounds__(64) void readout_kernel(
    const float* __restrict__ gq, const float* __restrict__ gi,
    const f16* __restrict__ w1, const float* __restrict__ b1,
    const f16* __restrict__ w2, const float* __restrict__ b2,
    const float* __restrict__ w3, const float* __restrict__ b3,
    float* __restrict__ out)
{
  __shared__ __align__(16) f16 smsg[RO_WAVES][16 * 256];
  __shared__ __align__(16) f16 sact1[RO_WAVES][16 * 128];
  __shared__ __align__(16) f16 sact2[RO_WAVES][16 * 128];

  const int wave = threadIdx.x >> 5;
  const int lane = threadIdx.x & 31;
  const int tile = blockIdx.x * RO_WAVES + wave;
  const int r0 = tile * 16;
  if (r0 >= SE * SE) return;

  f16* msg  = smsg[wave];
  f16* act1 = sact1[wave];
  f16* act2 = sact2[wave];

  for (int r = 0; r < 16; ++r) {
    int row = r0 + r;
    int i = row >> 8, j = row & 255;
    const float* pq = gq + (size_t)j * 128;
    const float* pi = gi + (size_t)i * 128;
    for (int k = lane; k < 256; k += 32)
      msg[r * 256 + k] = (f16)((k < 128) ? pq[k] : pi[k - 128]);
  }
  lds_fence_wave();

  const int nlo = lane & 15;
  const int mh  = (lane >> 4) << 3;

  for (int nt = 0; nt < 8; ++nt) {
    v8f acc = {};
    for (int kc = 0; kc < 256; kc += 32)
      acc = WMMA_F16(frag_a_lds(msg, 256, kc, lane),
                     frag_b_glb(w1, 256, nt * 16, kc, lane), acc);
    int n = nt * 16 + nlo;
    float bias = b1[n];
#pragma unroll
    for (int r = 0; r < 8; ++r)
      act1[(mh + r) * 128 + n] = (f16)fmaxf(acc[r] + bias, 0.0f);
  }
  lds_fence_wave();
  for (int nt = 0; nt < 8; ++nt) {
    v8f acc = {};
    for (int kc = 0; kc < 128; kc += 32)
      acc = WMMA_F16(frag_a_lds(act1, 128, kc, lane),
                     frag_b_glb(w2, 128, nt * 16, kc, lane), acc);
    int n = nt * 16 + nlo;
    float bias = b2[n];
#pragma unroll
    for (int r = 0; r < 8; ++r)
      act2[(mh + r) * 128 + n] = (f16)fmaxf(acc[r] + bias, 0.0f);
  }
  lds_fence_wave();
  if (lane < 16) {                        // final 128 -> 1 dot per row
    float acc = b3[0];
    for (int k = 0; k < 128; ++k) acc += (float)act2[lane * 128 + k] * w3[k];
    out[r0 + lane] = acc;
  }
}

// ---------------------------------------------------------------------------
// Host launcher
// ---------------------------------------------------------------------------

static inline int ceil_div(long long a, long long b) { return (int)((a + b - 1) / b); }

extern "C" void kernel_launch(void* const* d_in, const int* in_sizes, int n_in,
                              void* d_out, int out_size, void* d_ws, size_t ws_size,
                              hipStream_t stream) {
  (void)n_in; (void)out_size; (void)ws_size;

  const float* hs_in   = (const float*)d_in[0];
  const float* hf_in   = (const float*)d_in[1];
  const float* hseq_in = (const float*)d_in[2];
  const int*   esrc    = (const int*)d_in[3];
  const int*   edst    = (const int*)d_in[4];
  const int*   gtype   = (const int*)d_in[5];
  const int*   ffidx   = (const int*)d_in[6];
  const int E   = in_sizes[3];
  const int Nn  = in_sizes[5];
  const int Mff = in_sizes[6];

  // ---- param pointer map (setup_inputs insertion order, recursively) ----
  const float* aggrW[3][3][3]; const float* aggrB[3][3][3];
  const float* gWih[3][3]; const float* gWhh[3][3];
  const float* gBih[3][3]; const float* gBhh[3][3];
  int ip = 7;
  for (int g = 0; g < 3; ++g)
    for (int s = 0; s < 3; ++s) {
      for (int l = 0; l < 3; ++l) {
        aggrW[g][s][l] = (const float*)d_in[ip++];
        aggrB[g][s][l] = (const float*)d_in[ip++];
      }
      gWih[g][s] = (const float*)d_in[ip++];
      gWhh[g][s] = (const float*)d_in[ip++];
      gBih[g][s] = (const float*)d_in[ip++];
      gBhh[g][s] = (const float*)d_in[ip++];
    }
  const float* poolW[4][4];                    // pq, pi, pq2, pi2 : Wq,Wk,Wv,Wo
  for (int p = 0; p < 4; ++p)
    for (int w = 0; w < 4; ++w) poolW[p][w] = (const float*)d_in[ip++];
  const float* roW[2][3]; const float* roB[2][3];
  for (int rdx = 0; rdx < 2; ++rdx)
    for (int l = 0; l < 3; ++l) {
      roW[rdx][l] = (const float*)d_in[ip++];
      roB[rdx][l] = (const float*)d_in[ip++];
    }

  // ---- workspace carving ----
  char* wsp = (char*)d_ws;
  size_t off = 0;
  auto alloc = [&](size_t bytes) -> void* {
    off = (off + 255) & ~(size_t)255;
    void* p = wsp + off;
    off += bytes;
    return p;
  };
  const size_t hbytes = (size_t)Nn * H2 * sizeof(float);
  float* hsw   = (float*)alloc(hbytes);
  float* hfw   = (float*)alloc(hbytes);
  float* hseqw = (float*)alloc(hbytes);
  float* agg0  = (float*)alloc(hbytes);
  float* agg1  = (float*)alloc(hbytes);
  float* agg2  = (float*)alloc(hbytes);
  int* sorted_e = (int*)alloc((size_t)E * 4);
  int* sorted_n = (int*)alloc((size_t)Nn * 4);
  int* ctr      = (int*)alloc(64 * 4);   // [0..2] ecnt [3..5] ecur [6..8] ncnt [9..11] ncur

  const int KinTab[3] = {128, 256, 384};
  f16* aW1[3][3]; f16* aW2[3][3]; f16* aW3[3][3];
  f16* gWih16[3][3]; f16* gWhh16[3][3];
  for (int g = 0; g < 3; ++g)
    for (int s = 0; s < 3; ++s) {
      aW1[g][s]    = (f16*)alloc((size_t)128 * KinTab[s] * 2);
      aW2[g][s]    = (f16*)alloc((size_t)128 * 128 * 2);
      aW3[g][s]    = (f16*)alloc((size_t)128 * 128 * 2);
      gWih16[g][s] = (f16*)alloc((size_t)384 * 128 * 2);
      gWhh16[g][s] = (f16*)alloc((size_t)384 * 128 * 2);
    }
  f16* roW1T[2]; f16* roW2T[2];
  for (int rdx = 0; rdx < 2; ++rdx) {
    roW1T[rdx] = (f16*)alloc((size_t)128 * 256 * 2);
    roW2T[rdx] = (f16*)alloc((size_t)128 * 128 * 2);
  }
  float* gpool = (float*)alloc((size_t)4 * SE * 128 * 4);  // gq, gi, gq2, gi2
  float* gq  = gpool;
  float* gi  = gpool + (size_t)SE * 128;
  float* gq2 = gpool + (size_t)2 * SE * 128;
  float* gi2 = gpool + (size_t)3 * SE * 128;
  float* ffbuf = (float*)alloc((size_t)4 * Mff * 128 * 4); // h0,h1,h0t,h1t
  float* h0  = ffbuf;
  float* h1  = ffbuf + (size_t)Mff * 128;
  float* h0t = ffbuf + (size_t)2 * Mff * 128;
  float* h1t = ffbuf + (size_t)3 * Mff * 128;

  // ---- working copies of node states (never mutate inputs) ----
  hipMemcpyAsync(hsw, hs_in, hbytes, hipMemcpyDeviceToDevice, stream);
  hipMemcpyAsync(hfw, hf_in, hbytes, hipMemcpyDeviceToDevice, stream);
  hipMemcpyAsync(hseqw, hseq_in, hbytes, hipMemcpyDeviceToDevice, stream);

  // ---- bucket-sort edges and nodes by gate type ----
  zero_ints_kernel<<<1, 64, 0, stream>>>(ctr, 16);
  count_types_kernel<<<ceil_div(E, 256), 256, 0, stream>>>(edst, gtype, E, ctr + 0);
  count_types_kernel<<<ceil_div(Nn, 256), 256, 0, stream>>>(nullptr, gtype, Nn, ctr + 6);
  make_offsets_kernel<<<1, 1, 0, stream>>>(ctr + 0, ctr + 3);
  make_offsets_kernel<<<1, 1, 0, stream>>>(ctr + 6, ctr + 9);
  scatter_sorted_kernel<<<ceil_div(E, 256), 256, 0, stream>>>(edst, gtype, E, ctr + 3, sorted_e);
  scatter_sorted_kernel<<<ceil_div(Nn, 256), 256, 0, stream>>>(nullptr, gtype, Nn, ctr + 9, sorted_n);

  // ---- convert weights to f16 [out][in] ----
  for (int g = 0; g < 3; ++g)
    for (int s = 0; s < 3; ++s) {
      int K = KinTab[s];
      convert_transpose_kernel<<<ceil_div((long long)K * 128, 256), 256, 0, stream>>>(
          aggrW[g][s][0], aW1[g][s], K, 128);
      convert_transpose_kernel<<<ceil_div(128 * 128, 256), 256, 0, stream>>>(
          aggrW[g][s][1], aW2[g][s], 128, 128);
      convert_transpose_kernel<<<ceil_div(128 * 128, 256), 256, 0, stream>>>(
          aggrW[g][s][2], aW3[g][s], 128, 128);
      convert_plain_kernel<<<ceil_div(384 * 128, 256), 256, 0, stream>>>(
          gWih[g][s], gWih16[g][s], 384 * 128);
      convert_plain_kernel<<<ceil_div(384 * 128, 256), 256, 0, stream>>>(
          gWhh[g][s], gWhh16[g][s], 384 * 128);
    }
  for (int rdx = 0; rdx < 2; ++rdx) {
    convert_transpose_kernel<<<ceil_div(256 * 128, 256), 256, 0, stream>>>(
        roW[rdx][0], roW1T[rdx], 256, 128);
    convert_transpose_kernel<<<ceil_div(128 * 128, 256), 256, 0, stream>>>(
        roW[rdx][1], roW2T[rdx], 128, 128);
  }

  // ---- message-passing rounds ----
  const int etiles = ceil_div(E, 16);
  const int ntiles = ceil_div(Nn, 16);
  const int eblocks = ceil_div(etiles, EW_WAVES);
  const int nblocks = ceil_div(ntiles, GR_WAVES);
  float* aggs[3] = {agg0, agg1, agg2};

  for (int round = 0; round < 2; ++round) {
    // all aggregations first (messages use pre-round states), then all GRUs
    for (int s = 0; s < 3; ++s) {
      zero_floats_kernel<<<ceil_div((long long)Nn * H2, 256), 256, 0, stream>>>(
          aggs[s], (long long)Nn * H2);
      BranchW bw;
      for (int g = 0; g < 3; ++g) {
        bw.w1[g] = aW1[g][s]; bw.w2[g] = aW2[g][s]; bw.w3[g] = aW3[g][s];
        bw.b1[g] = aggrB[g][s][0]; bw.b2[g] = aggrB[g][s][1]; bw.b3[g] = aggrB[g][s][2];
      }
      edge_mlp_kernel<<<eblocks, 64, 0, stream>>>(
          sorted_e, E, esrc, edst, gtype, hsw, hfw, hseqw, KinTab[s], bw, aggs[s]);
    }
    for (int s = 0; s < 3; ++s) {
      GruW gw;
      for (int g = 0; g < 3; ++g) {
        gw.wih[g] = gWih16[g][s]; gw.whh[g] = gWhh16[g][s];
        gw.bih[g] = gBih[g][s];   gw.bhh[g] = gBhh[g][s];
      }
      float* hb = (s == 0) ? hsw : (s == 1) ? hfw : hseqw;
      gru_kernel<<<nblocks, 128, 0, stream>>>(sorted_n, Nn, gtype, aggs[s], hb, gw);
    }
  }

  // ---- readout ----
  gather_ff_kernel<<<ceil_div((long long)Mff * 128, 128), 128, 0, stream>>>(
      hsw, hfw, hseqw, ffidx, Mff, h0, h1, h0t, h1t);
  pool_kernel<<<SE, 128, 0, stream>>>(h0, h1, Mff,
      poolW[0][0], poolW[0][1], poolW[0][2], poolW[0][3], gq);
  pool_kernel<<<SE, 128, 0, stream>>>(h0, h1, Mff,
      poolW[1][0], poolW[1][1], poolW[1][2], poolW[1][3], gi);
  pool_kernel<<<SE, 128, 0, stream>>>(h0t, h1t, Mff,
      poolW[2][0], poolW[2][1], poolW[2][2], poolW[2][3], gq2);
  pool_kernel<<<SE, 128, 0, stream>>>(h0t, h1t, Mff,
      poolW[3][0], poolW[3][1], poolW[3][2], poolW[3][3], gi2);

  float* outp = (float*)d_out;
  const int roblocks = ceil_div(SE * SE / 16, RO_WAVES);
  readout_kernel<<<roblocks, 64, 0, stream>>>(
      gq, gi, roW1T[0], roB[0][0], roW2T[0], roB[0][1], roW[0][2], roB[0][2], outp);
  readout_kernel<<<roblocks, 64, 0, stream>>>(
      gq2, gi2, roW1T[1], roB[1][0], roW2T[1], roB[1][1], roW[1][2], roB[1][2],
      outp + SE * SE);
}
